// PurifiedVAE_84095459656129
// MI455X (gfx1250) — matrix operand order repaired and checked by reading, not compiled, not run on hardware
//
#include <hip/hip_runtime.h>
#include <hip/hip_bf16.h>
#include <stdint.h>

// ---------------- problem constants ----------------
#define BB   256   // batch
#define TT   512   // timesteps
#define II   12    // input dims
#define HH   1024  // hidden
#define ZZ   128   // latent
#define H3   3072  // 3*HH

#define APAD 1032  // LDS A-panel row stride in bf16 elems (1024 + 8 pad -> 2064B, bank-rotating)

typedef __attribute__((ext_vector_type(16))) __bf16 v16bf;
typedef __attribute__((ext_vector_type(8)))  float  v8f;

union FragBF { v16bf v; uint4 u[2]; };

__device__ __forceinline__ float sigf(float x) {
    return 1.0f / (1.0f + __expf(-x));
}

__device__ __forceinline__ void async_g2lds_b128(const void* gsrc, unsigned lds_off) {
    // GLOBAL_LOAD_ASYNC_TO_LDS_B128: VDST = LDS byte offset VGPR, ADDR = 64-bit global addr
    asm volatile("global_load_async_to_lds_b128 %0, %1, off"
                 :: "v"(lds_off), "v"(gsrc) : "memory");
}

__device__ __forceinline__ void wait_async0() {
#if __has_builtin(__builtin_amdgcn_s_wait_asynccnt)
    __builtin_amdgcn_s_wait_asynccnt(0);
#else
    asm volatile("s_wait_asynccnt 0x0" ::: "memory");
#endif
}

// ---------------- utility kernels ----------------
__global__ void f2bf_kernel(const float* __restrict__ src,
                            __hip_bfloat16* __restrict__ dst, int n) {
    int i = blockIdx.x * blockDim.x + threadIdx.x;
    if (i < n) dst[i] = __float2bfloat16(src[i]);
}

__global__ void zero32_kernel(uint32_t* __restrict__ p, int n) {
    int i = blockIdx.x * blockDim.x + threadIdx.x;
    if (i < n) p[i] = 0u;
}

// ---------------- fused GRU step (WMMA bf16, fp32 accum) ----------------
// gh = h @ Whh.T via v_wmma_f32_16x16x32_bf16.
// Block = 8 waves sharing one 16-row batch tile. The shared A panel
// (16x1024 bf16, 32KB) is staged once into LDS with async global->LDS
// copies (ASYNCcnt); fragments are then read with ds_load_b128.
// Each wave owns one 16-column hidden tile and issues 3 WMMAs per K-chunk
// (r/z/n gate row groups of Whh). B operand is double-buffered in VGPRs.
// Epilogue fuses the K=12 input projection, gate math, state update and
// bf16 repack of h for the next step's A operand.
template<bool DEC>
__global__ __launch_bounds__(256)
void gru_step_kernel(const __hip_bfloat16* __restrict__ hbf,   // [B,H] bf16 (A)
                     const float*          __restrict__ hf,    // [B,H] f32
                     const __hip_bfloat16* __restrict__ whh,   // [3H,H] bf16
                     const float*          __restrict__ x,     // [B,T,I]
                     const float*          __restrict__ wih,   // [3H,ldw]
                     int ldw,
                     const float*          __restrict__ bih,   // [3H] (enc only)
                     const float*          __restrict__ bhh,   // [3H]
                     const float*          __restrict__ gzbuf, // [B,3H] (dec only)
                     int tstep,
                     float*          __restrict__ hf_out,
                     __hip_bfloat16* __restrict__ hbf_out) {
    __shared__ __hip_bfloat16 sA[16 * APAD];

    const int lane  = threadIdx.x & 31;
    const int mtile = blockIdx.x >> 3;                              // 0..15 (uniform per block)
    const int jtile = ((blockIdx.x & 7) << 3) + (threadIdx.x >> 5); // 0..63

    // ---- stage A panel (16 x 1024 bf16) into LDS, async ----
    {
        const int tid = threadIdx.x;
        #pragma unroll
        for (int it = 0; it < 8; ++it) {
            const int chunk = tid + (it << 8);   // 0..2047 (16B units)
            const int row   = chunk >> 7;        // 0..15
            const int kc    = chunk & 127;       // 0..127
            const __hip_bfloat16* g =
                hbf + (size_t)((mtile << 4) + row) * HH + (size_t)kc * 8;
            const unsigned ldst =
                (unsigned)(size_t)(sA + row * APAD + kc * 8);
            async_g2lds_b128(g, ldst);
        }
        wait_async0();
        __syncthreads();
    }

    const int l15     = lane & 15;
    const int half8   = (lane >> 4) << 3;   // A: 0 or 8
    const int khalf16 = (lane >> 4) << 4;   // 0 or 16, unused (B uses half8)
    const int jcol    = (jtile << 4) + l15; // output hidden column

    const __hip_bfloat16* arow   = sA + l15 * APAD;          // LDS
    const __hip_bfloat16* br_row = whh + (size_t)jcol * HH;
    const __hip_bfloat16* bz_row = whh + (size_t)(HH + jcol) * HH;
    const __hip_bfloat16* bn_row = whh + (size_t)(2 * HH + jcol) * HH;

    v8f cr = {0.f,0.f,0.f,0.f,0.f,0.f,0.f,0.f};
    v8f cz = {0.f,0.f,0.f,0.f,0.f,0.f,0.f,0.f};
    v8f cn = {0.f,0.f,0.f,0.f,0.f,0.f,0.f,0.f};

    // prefetch fp32 h rows needed by the epilogue
    const int mbase = (mtile << 4) + ((lane >> 4) << 3);
    __builtin_prefetch(hf + (size_t)mbase * HH + jcol, 0, 3);

    auto loadB = [&](FragBF* f, int k0) {
        const uint4* p;
        p = (const uint4*)(br_row + k0 + half8); f[0].u[0] = p[0]; f[0].u[1] = p[2];
        p = (const uint4*)(bz_row + k0 + half8); f[1].u[0] = p[0]; f[1].u[1] = p[2];
        p = (const uint4*)(bn_row + k0 + half8); f[2].u[0] = p[0]; f[2].u[1] = p[2];
    };
    auto loadA = [&](FragBF& a, int k0) {
        // A (16x32 bf16): lanes 0-15 K={0..7,16..23}, lanes 16-31 K={8..15,24..31}
        a.u[0] = *(const uint4*)(arow + k0 + half8);        // ds_load_b128
        a.u[1] = *(const uint4*)(arow + k0 + half8 + 16);   // ds_load_b128
    };

    FragBF b0[3], b1[3], a;
    loadB(b0, 0);
    #pragma unroll 4
    for (int k0 = 0; k0 < HH; k0 += 64) {
        loadB(b1, k0 + 32);
        loadA(a, k0);
        cr = __builtin_amdgcn_wmma_f32_16x16x32_bf16(false, a.v, false, b0[0].v,
                                                     (short)0, cr, false, false);
        cz = __builtin_amdgcn_wmma_f32_16x16x32_bf16(false, a.v, false, b0[1].v,
                                                     (short)0, cz, false, false);
        cn = __builtin_amdgcn_wmma_f32_16x16x32_bf16(false, a.v, false, b0[2].v,
                                                     (short)0, cn, false, false);
        if (k0 + 64 < HH) loadB(b0, k0 + 64);
        loadA(a, k0 + 32);
        cr = __builtin_amdgcn_wmma_f32_16x16x32_bf16(false, a.v, false, b1[0].v,
                                                     (short)0, cr, false, false);
        cz = __builtin_amdgcn_wmma_f32_16x16x32_bf16(false, a.v, false, b1[1].v,
                                                     (short)0, cz, false, false);
        cn = __builtin_amdgcn_wmma_f32_16x16x32_bf16(false, a.v, false, b1[2].v,
                                                     (short)0, cn, false, false);
    }

    // ---- epilogue: gates for 8 batch rows (C layout: VGPR i -> M = i or 8+i) ----
    const float bhr = bhh[jcol];
    const float bhz = bhh[HH + jcol];
    const float bhn = bhh[2 * HH + jcol];
    float bir = 0.f, biz = 0.f, bin_ = 0.f;
    if (!DEC) {
        bir  = bih[jcol];
        biz  = bih[HH + jcol];
        bin_ = bih[2 * HH + jcol];
    }

    float wr[II], wz[II], wn[II];
    #pragma unroll
    for (int k = 0; k < II; ++k) {
        wr[k] = wih[(size_t)jcol * ldw + k];
        wz[k] = wih[(size_t)(HH + jcol) * ldw + k];
        wn[k] = wih[(size_t)(2 * HH + jcol) * ldw + k];
    }

    #pragma unroll
    for (int i = 0; i < 8; ++i) {
        const int m = mbase + i;

        float xv[II];
        if (DEC && tstep == 0) {
            #pragma unroll
            for (int k = 0; k < II; ++k) xv[k] = (k == II - 1) ? 1.0f : 0.0f;
        } else {
            const int xt = DEC ? (tstep - 1) : tstep;
            const float* xr = x + (size_t)m * (TT * II) + (size_t)xt * II;
            #pragma unroll
            for (int k = 0; k < II; ++k) xv[k] = xr[k];
        }

        float gr, gz2, gn;
        if (DEC) {
            const float* g = gzbuf + (size_t)m * H3;
            gr  = g[jcol];
            gz2 = g[HH + jcol];
            gn  = g[2 * HH + jcol];
        } else {
            gr = bir; gz2 = biz; gn = bin_;
        }
        #pragma unroll
        for (int k = 0; k < II; ++k) {
            gr  = fmaf(xv[k], wr[k], gr);
            gz2 = fmaf(xv[k], wz[k], gz2);
            gn  = fmaf(xv[k], wn[k], gn);
        }

        const float hr = cr[i] + bhr;
        const float hz = cz[i] + bhz;
        const float hn = cn[i] + bhn;

        const float r  = sigf(gr + hr);
        const float zg = sigf(gz2 + hz);
        const float nn = tanhf(gn + r * hn);
        const float ho = hf[(size_t)m * HH + jcol];
        const float hnew = (1.0f - zg) * nn + zg * ho;

        hf_out[(size_t)m * HH + jcol]  = hnew;
        hbf_out[(size_t)m * HH + jcol] = __float2bfloat16(hnew);
    }
}

// ---------------- VAE heads: mean / stddev / z ----------------
__global__ __launch_bounds__(256)
void head_kernel(const float* __restrict__ hfF, const float* __restrict__ hfB,
                 const float* __restrict__ Wmu, const float* __restrict__ bmu,
                 const float* __restrict__ Wvar, const float* __restrict__ bvar,
                 const float* __restrict__ noise,
                 float* __restrict__ mean_out, float* __restrict__ std_out,
                 float* __restrict__ z_out) {
    const int tid = blockIdx.x * 256 + threadIdx.x;  // B*Z
    const int b = tid >> 7, zc = tid & 127;
    const float4* h1 = (const float4*)(hfF + (size_t)b * HH);
    const float4* h2 = (const float4*)(hfB + (size_t)b * HH);
    const float4* wm1 = (const float4*)(Wmu + (size_t)zc * 2 * HH);
    const float4* wm2 = (const float4*)(Wmu + (size_t)zc * 2 * HH + HH);
    const float4* wv1 = (const float4*)(Wvar + (size_t)zc * 2 * HH);
    const float4* wv2 = (const float4*)(Wvar + (size_t)zc * 2 * HH + HH);
    float mu = 0.f, lv = 0.f;
    for (int k = 0; k < HH / 4; ++k) {
        const float4 a = h1[k], c = h2[k];
        float4 w = wm1[k];
        mu = fmaf(a.x, w.x, fmaf(a.y, w.y, fmaf(a.z, w.z, fmaf(a.w, w.w, mu))));
        w = wm2[k];
        mu = fmaf(c.x, w.x, fmaf(c.y, w.y, fmaf(c.z, w.z, fmaf(c.w, w.w, mu))));
        w = wv1[k];
        lv = fmaf(a.x, w.x, fmaf(a.y, w.y, fmaf(a.z, w.z, fmaf(a.w, w.w, lv))));
        w = wv2[k];
        lv = fmaf(c.x, w.x, fmaf(c.y, w.y, fmaf(c.z, w.z, fmaf(c.w, w.w, lv))));
    }
    const float mean = mu + bmu[zc];
    const float sd   = __expf(0.5f * (lv + bvar[zc]));
    mean_out[tid] = mean;
    std_out[tid]  = sd;
    z_out[tid]    = mean + sd * noise[tid];
}

// ---------------- decoder init state: h = tanh(z @ W_init.T + b) ----------------
__global__ __launch_bounds__(256)
void hx_kernel(const float* __restrict__ z, const float* __restrict__ Winit,
               const float* __restrict__ binit,
               float* __restrict__ hf_out, __hip_bfloat16* __restrict__ hbf_out) {
    const int tid = blockIdx.x * 256 + threadIdx.x;  // B*H
    const int b = tid >> 10, j = tid & 1023;
    const float4* zp = (const float4*)(z + (size_t)b * ZZ);
    const float4* wp = (const float4*)(Winit + (size_t)j * ZZ);
    float acc = 0.f;
    #pragma unroll 4
    for (int k = 0; k < ZZ / 4; ++k) {
        const float4 a = zp[k], w = wp[k];
        acc = fmaf(a.x, w.x, fmaf(a.y, w.y, fmaf(a.z, w.z, fmaf(a.w, w.w, acc))));
    }
    const float h = tanhf(acc + binit[j]);
    hf_out[tid]  = h;
    hbf_out[tid] = __float2bfloat16(h);
}

// ---------------- decoder z-projection precompute: gz = z @ WcIh[:,12:].T + bc_ih ----------------
__global__ __launch_bounds__(256)
void gz_kernel(const float* __restrict__ z, const float* __restrict__ WcIh,
               const float* __restrict__ bcIh, float* __restrict__ gz_out) {
    const int tid = blockIdx.x * 256 + threadIdx.x;  // B*3H
    const int b = tid / H3, c = tid - b * H3;
    const float4* zp = (const float4*)(z + (size_t)b * ZZ);
    const float4* wp = (const float4*)(WcIh + (size_t)c * (II + ZZ) + II);
    float acc = bcIh[c];
    #pragma unroll 4
    for (int k = 0; k < ZZ / 4; ++k) {
        const float4 a = zp[k], w = wp[k];
        acc = fmaf(a.x, w.x, fmaf(a.y, w.y, fmaf(a.z, w.z, fmaf(a.w, w.w, acc))));
    }
    gz_out[tid] = acc;
}

// ---------------- per-step output projection: recon = sigmoid(h @ W_out.T + b) ----------------
__global__ __launch_bounds__(256)
void outproj_kernel(const float* __restrict__ hf, const float* __restrict__ Wout,
                    const float* __restrict__ bout, float* __restrict__ recon,
                    int tstep) {
    const int tid = blockIdx.x * 256 + threadIdx.x;  // B*I
    const int b = tid / II, i = tid - b * II;
    const float4* hp = (const float4*)(hf + (size_t)b * HH);
    const float4* wp = (const float4*)(Wout + (size_t)i * HH);
    float acc = 0.f;
    for (int k = 0; k < HH / 4; ++k) {
        const float4 a = hp[k], w = wp[k];
        acc = fmaf(a.x, w.x, fmaf(a.y, w.y, fmaf(a.z, w.z, fmaf(a.w, w.w, acc))));
    }
    recon[(size_t)b * TT * II + (size_t)tstep * II + i] = sigf(acc + bout[i]);
}

// ---------------- host orchestration ----------------
extern "C" void kernel_launch(void* const* d_in, const int* in_sizes, int n_in,
                              void* d_out, int out_size, void* d_ws, size_t ws_size,
                              hipStream_t stream) {
    (void)in_sizes; (void)n_in; (void)out_size;
    const float* x     = (const float*)d_in[0];
    const float* noise = (const float*)d_in[1];
    const float* WihF  = (const float*)d_in[2];
    const float* WhhF  = (const float*)d_in[3];
    const float* bihF  = (const float*)d_in[4];
    const float* bhhF  = (const float*)d_in[5];
    const float* WihB  = (const float*)d_in[6];
    const float* WhhB  = (const float*)d_in[7];
    const float* bihB  = (const float*)d_in[8];
    const float* bhhB  = (const float*)d_in[9];
    const float* Wmu   = (const float*)d_in[10];
    const float* bmu   = (const float*)d_in[11];
    const float* Wvar  = (const float*)d_in[12];
    const float* bvar  = (const float*)d_in[13];
    const float* Winit = (const float*)d_in[14];
    const float* binit = (const float*)d_in[15];
    const float* Wout  = (const float*)d_in[16];
    const float* bout  = (const float*)d_in[17];
    const float* WcIh  = (const float*)d_in[18];
    const float* WcHh  = (const float*)d_in[19];
    const float* bcIh  = (const float*)d_in[20];
    const float* bcHh  = (const float*)d_in[21];

    // ---- workspace layout ----
    const size_t SZ_WBF  = (size_t)H3 * HH * sizeof(__hip_bfloat16); // 6 MB
    const size_t SZ_HF   = (size_t)BB * HH * sizeof(float);          // 1 MB
    const size_t SZ_HBF  = (size_t)BB * HH * sizeof(__hip_bfloat16); // 0.5 MB
    const size_t DIRBLK  = 2 * SZ_HF + 2 * SZ_HBF;                   // 3 MB
    const size_t OFF_H   = 3 * SZ_WBF;
    const size_t OFF_Z   = OFF_H + 3 * DIRBLK;
    const size_t OFF_GZ  = OFF_Z + (size_t)BB * ZZ * sizeof(float);
    const size_t TOTAL   = OFF_GZ + (size_t)BB * H3 * sizeof(float);
    if (ws_size < TOTAL) return;

    uint8_t* ws = (uint8_t*)d_ws;
    __hip_bfloat16* whhF_bf = (__hip_bfloat16*)(ws);
    __hip_bfloat16* whhB_bf = (__hip_bfloat16*)(ws + SZ_WBF);
    __hip_bfloat16* wcHh_bf = (__hip_bfloat16*)(ws + 2 * SZ_WBF);

    float*          hf32[3][2];
    __hip_bfloat16* hbf[3][2];
    for (int d = 0; d < 3; ++d) {
        uint8_t* base = ws + OFF_H + (size_t)d * DIRBLK;
        hf32[d][0] = (float*)(base);
        hf32[d][1] = (float*)(base + SZ_HF);
        hbf[d][0]  = (__hip_bfloat16*)(base + 2 * SZ_HF);
        hbf[d][1]  = (__hip_bfloat16*)(base + 2 * SZ_HF + SZ_HBF);
    }
    float* z_ws  = (float*)(ws + OFF_Z);
    float* gz_ws = (float*)(ws + OFF_GZ);

    float* recon    = (float*)d_out;                     // [B,T,I]
    float* mean_out = recon + (size_t)BB * TT * II;      // [B,Z]
    float* std_out  = mean_out + (size_t)BB * ZZ;        // [B,Z]

    // ---- pack recurrent weights to bf16 ----
    const int nW = H3 * HH;
    f2bf_kernel<<<nW / 256, 256, 0, stream>>>(WhhF, whhF_bf, nW);
    f2bf_kernel<<<nW / 256, 256, 0, stream>>>(WhhB, whhB_bf, nW);
    f2bf_kernel<<<nW / 256, 256, 0, stream>>>(WcHh, wcHh_bf, nW);

    // ---- zero h0 (fp32 + bf16) for both encoder directions ----
    zero32_kernel<<<(int)(SZ_HF / 4 / 256), 256, 0, stream>>>((uint32_t*)hf32[0][0], (int)(SZ_HF / 4));
    zero32_kernel<<<(int)(SZ_HBF / 4 / 256), 256, 0, stream>>>((uint32_t*)hbf[0][0], (int)(SZ_HBF / 4));
    zero32_kernel<<<(int)(SZ_HF / 4 / 256), 256, 0, stream>>>((uint32_t*)hf32[1][0], (int)(SZ_HF / 4));
    zero32_kernel<<<(int)(SZ_HBF / 4 / 256), 256, 0, stream>>>((uint32_t*)hbf[1][0], (int)(SZ_HBF / 4));

    // ---- encoder: forward direction ----
    for (int t = 0; t < TT; ++t) {
        gru_step_kernel<false><<<128, 256, 0, stream>>>(
            hbf[0][t & 1], hf32[0][t & 1], whhF_bf, x, WihF, II,
            bihF, bhhF, nullptr, t, hf32[0][(t + 1) & 1], hbf[0][(t + 1) & 1]);
    }
    // ---- encoder: backward direction (x time index reversed) ----
    for (int t = 0; t < TT; ++t) {
        gru_step_kernel<false><<<128, 256, 0, stream>>>(
            hbf[1][t & 1], hf32[1][t & 1], whhB_bf, x, WihB, II,
            bihB, bhhB, nullptr, TT - 1 - t, hf32[1][(t + 1) & 1], hbf[1][(t + 1) & 1]);
    }

    // final hidden states land in buffer 0 (512 steps, even)
    head_kernel<<<(BB * ZZ) / 256, 256, 0, stream>>>(
        hf32[0][0], hf32[1][0], Wmu, bmu, Wvar, bvar, noise,
        mean_out, std_out, z_ws);

    // ---- decoder init + z projection ----
    hx_kernel<<<(BB * HH) / 256, 256, 0, stream>>>(z_ws, Winit, binit,
                                                   hf32[2][0], hbf[2][0]);
    gz_kernel<<<(BB * H3) / 256, 256, 0, stream>>>(z_ws, WcIh, bcIh, gz_ws);

    // ---- decoder: teacher-forced GRUCell + output projection ----
    for (int t = 0; t < TT; ++t) {
        gru_step_kernel<true><<<128, 256, 0, stream>>>(
            hbf[2][t & 1], hf32[2][t & 1], wcHh_bf, x, WcIh, II + ZZ,
            nullptr, bcHh, gz_ws, t, hf32[2][(t + 1) & 1], hbf[2][(t + 1) & 1]);
        outproj_kernel<<<(BB * II) / 256, 256, 0, stream>>>(
            hf32[2][(t + 1) & 1], Wout, bout, recon, t);
    }
}
